// SelfConsistNet_37185826849257
// MI455X (gfx1250) — matrix operand order repaired and checked by reading, not compiled, not run on hardware
//
#include <hip/hip_runtime.h>
#include <hip/hip_bf16.h>
#include <math.h>

// =====================================================================
// SelfConsistNet on MI455X (gfx1250), wave32 + WMMA bf16 implicit GEMM.
//
// v4: all intermediate activations stored as bf16 in HBM -> halves the
// dominant inter-layer traffic (~2 GB -> ~1 GB against 23.3 TB/s) and
// turns conv B-staging into straight u16 moves (no cvt in hot path).
// Conv math stays on v_wmma_f32_16x16x32_bf16 with f32 accumulate;
// BN/residual/ReLU epilogue in f32, bf16 store.
//
// TDM / async-to-LDS evaluated and rejected for this workload: A-staging
// is a type-converting load (f32 weights -> bf16 LDS) and B-staging is a
// 2-byte-granularity im2col gather; neither matches TDM tile moves or
// B8/B32+ async granularity. Data path: global_prefetch_b8 + cooperative
// LDS staging + WMMA.
// =====================================================================

typedef __bf16 bf16_t;
typedef __attribute__((ext_vector_type(16))) __bf16 v16bf;
typedef __attribute__((ext_vector_type(8)))  float  v8f;

// ---------------------------------------------------------------------
// conv + BN (or bias-only) + residual + ReLU, implicit GEMM.
// grid = (P/64, ceil(Cout/32), B), block = 128 (4 waves).
// Block tile: 32 Cout x 64 pixels; each wave owns one 16-pixel subtile
// and both 16-row Cout subtiles (2 accumulators).
// TIN = float (stem reads the f32 image) or bf16_t (all other layers).
// ---------------------------------------------------------------------
template <int KH, int KW, int STRIDE, int PAD, typename TIN>
__global__ __launch_bounds__(128)
void conv_bn_wmma(const TIN* __restrict__ in, const float* __restrict__ wgt,
                  const float* __restrict__ bn_bias, const float* __restrict__ bn_mean,
                  const float* __restrict__ bn_scale, const float* __restrict__ bn_var,
                  const float* __restrict__ bias_only, const bf16_t* __restrict__ resid,
                  bf16_t* __restrict__ out,
                  int Cin, int H, int W, int Cout, int Ho, int Wo, int relu)
{
  constexpr int KHW = KH * KW;
  const int K = Cin * KHW;

  const int b     = blockIdx.z;
  const int mtile = blockIdx.y;        // 32 output channels
  const int ptile = blockIdx.x;        // 64 output pixels
  const int tid   = threadIdx.x;
  const int wave  = tid >> 5;
  const int lane  = tid & 31;
  const int half  = lane >> 4;
  const int nl    = lane & 15;

  __shared__ bf16_t sA[32][32];        // weights  [m][k]
  __shared__ bf16_t sB[32][64];        // activ.   [k][pixel-in-tile]

  // ---- staging coordinates: each thread owns one pixel column ----
  const int pixB = tid & 63;                    // column 0..63
  const int kPar = tid >> 6;                    // k parity (0/1)
  const int pS   = ptile * 64 + pixB;
  const int iy0  = (pS / Wo) * STRIDE - PAD;
  const int ix0  = (pS % Wo) * STRIDE - PAD;

  // ---- this lane's output pixel (fragments / epilogue) ----
  const int pO  = ptile * 64 + wave * 16 + nl;
  const int oyO = pO / Wo;
  const int oxO = pO % Wo;

  const TIN* inB = in + (size_t)b * Cin * H * W;

  v8f acc0 = {}, acc1 = {};

  for (int k0 = 0; k0 < K; k0 += 32) {
    __syncthreads();

    // ---- stage A: 32 x 32 weight tile (clamped, unconditional loads) ----
    #pragma unroll
    for (int e = 0; e < 8; ++e) {
      const int idx = tid + e * 128;            // 0..1023
      const int m   = idx >> 5;
      const int kk  = idx & 31;
      const int gm  = mtile * 32 + m;
      const int gk  = k0 + kk;
      const bool ok = (gk < K) & (gm < Cout);
      const int gmc = ok ? gm : 0;
      const int gkc = ok ? gk : 0;
      const float wv = wgt[(size_t)gmc * K + gkc];
      sA[m][kk] = (bf16_t)(ok ? wv : 0.f);
    }
    if (k0 + 32 < K)   // global_prefetch_b8 for the next weight chunk
      __builtin_prefetch(&wgt[(size_t)(mtile * 32) * K + k0 + 32], 0, 1);

    // ---- stage B: 32 x 64 activation tile, im2col with clamped loads ----
    bf16_t xv[16];
    #pragma unroll
    for (int t = 0; t < 16; ++t) {
      const int kl  = kPar + 2 * t;             // 0..31
      int gk        = k0 + kl;
      const bool kok = (gk < K);
      gk            = kok ? gk : (K - 1);
      const int cin = gk / KHW;                 // constant-divisor math
      const int rem = gk - cin * KHW;
      const int r   = rem / KW;
      const int s   = rem - r * KW;
      const int iy  = iy0 + r;
      const int ix  = ix0 + s;
      const bool ok = kok & ((unsigned)iy < (unsigned)H) & ((unsigned)ix < (unsigned)W);
      const int iyc = ok ? iy : 0;
      const int ixc = ok ? ix : 0;
      const bf16_t x = (bf16_t)inB[((size_t)cin * H + iyc) * W + ixc];
      xv[t] = ok ? x : (bf16_t)0.f;
    }
    #pragma unroll
    for (int t = 0; t < 16; ++t)
      sB[kPar + 2 * t][pixB] = xv[t];

    __syncthreads();

    // ---- fragments from LDS ----
    v16bf a0, a1, bfrag;
    #pragma unroll
    for (int v = 0; v < 8; ++v) {
      const int kk = ((v >> 2) << 4) + (half << 3) + ((v & 3) << 1);
      a0[2 * v]     = sA[nl][kk];
      a0[2 * v + 1] = sA[nl][kk + 1];
      a1[2 * v]     = sA[16 + nl][kk];
      a1[2 * v + 1] = sA[16 + nl][kk + 1];
    }
    const int pcol = wave * 16 + nl;
    #pragma unroll
    for (int v = 0; v < 8; ++v) {
      const int kk = ((v >> 2) << 4) + (half << 3) + ((v & 3) << 1);
      bfrag[2 * v]     = sB[kk][pcol];
      bfrag[2 * v + 1] = sB[kk + 1][pcol];
    }

    acc0 = __builtin_amdgcn_wmma_f32_16x16x32_bf16(false, a0, false, bfrag,
                                                   (short)0, acc0, false, false);
    acc1 = __builtin_amdgcn_wmma_f32_16x16x32_bf16(false, a1, false, bfrag,
                                                   (short)0, acc1, false, false);
  }

  // ---- epilogue: BN / bias, residual, ReLU; f32 math, bf16 store ----
  #pragma unroll
  for (int j = 0; j < 2; ++j) {
    const v8f& cacc = j ? acc1 : acc0;
    #pragma unroll
    for (int i = 0; i < 8; ++i) {
      const int cout = mtile * 32 + j * 16 + i + (half << 3);
      if (cout >= Cout) continue;               // only for Cout==16 (embed)
      float v = cacc[i];
      if (bias_only) {
        v += bias_only[cout];
      } else {
        const float inv = bn_scale[cout] * rsqrtf(bn_var[cout] + 1e-5f);
        v = (v - bn_mean[cout]) * inv + bn_bias[cout];
      }
      const size_t oidx = (((size_t)b * Cout + cout) * Ho + oyO) * Wo + oxO;
      if (resid) v += (float)resid[oidx];
      if (relu)  v = fmaxf(v, 0.f);
      out[oidx] = (bf16_t)v;
    }
  }
}

// ---------------------------------------------------------------------
// 3x3 stride-2 pad-1 max pool (stem), bf16 activations.
// ---------------------------------------------------------------------
__global__ void maxpool3x3s2(const bf16_t* __restrict__ in, bf16_t* __restrict__ out,
                             int B, int C, int H, int W, int Ho, int Wo)
{
  size_t idx = (size_t)blockIdx.x * blockDim.x + threadIdx.x;
  const size_t total = (size_t)B * C * Ho * Wo;
  if (idx >= total) return;
  int wo = (int)(idx % Wo); size_t r = idx / Wo;
  int ho = (int)(r % Ho);   r /= Ho;
  int c  = (int)(r % C);
  int b  = (int)(r / C);
  const bf16_t* base = in + (((size_t)b * C + c) * H) * W;
  float m = -INFINITY;
  #pragma unroll
  for (int dy = 0; dy < 3; ++dy)
    #pragma unroll
    for (int dx = 0; dx < 3; ++dx) {
      const int iy = ho * 2 - 1 + dy;
      const int ix = wo * 2 - 1 + dx;
      const bool ok = ((unsigned)iy < (unsigned)H) & ((unsigned)ix < (unsigned)W);
      const int iyc = ok ? iy : 0;
      const int ixc = ok ? ix : 0;
      const float v = (float)base[(size_t)iyc * W + ixc];
      m = fmaxf(m, ok ? v : -INFINITY);
    }
  out[idx] = (bf16_t)m;   // exact: m is one of the bf16 inputs
}

// ---------------------------------------------------------------------
// Global average pool over trailing 8x8 -> [B,512] f32.
// ---------------------------------------------------------------------
__global__ void gap8x8(const bf16_t* __restrict__ in, float* __restrict__ out, int BC)
{
  const int idx = blockIdx.x * blockDim.x + threadIdx.x;
  if (idx >= BC) return;
  const bf16_t* p = in + (size_t)idx * 64;
  float s = 0.f;
  #pragma unroll
  for (int i = 0; i < 64; ++i) s += (float)p[i];
  out[idx] = s * (1.f / 64.f);
}

// ---------------------------------------------------------------------
// Head: pred[b,cls] = sum_c pool[b,c] * head_w[cls,c]   (tiny, scalar)
// ---------------------------------------------------------------------
__global__ void head2(const float* __restrict__ pv, const float* __restrict__ hw,
                      float* __restrict__ pred, int B)
{
  const int idx = blockIdx.x * blockDim.x + threadIdx.x;
  if (idx >= B * 2) return;
  const int b = idx >> 1, cls = idx & 1;
  float s = 0.f;
  for (int c = 0; c < 512; ++c) s += pv[b * 512 + c] * hw[cls * 512 + c];
  pred[idx] = s;
}

// ---------------------------------------------------------------------
// Consistency volume: per batch, V = sigmoid((E^T E)/sqrt(16)).
// E = [16 chans, 256 positions] bf16; K=16 zero-padded to 32 -> 1 WMMA.
// K-padding resolved at compile time (A) / one uniform cndmask (B).
// ---------------------------------------------------------------------
__global__ __launch_bounds__(32)
void corr_sigmoid_wmma(const bf16_t* __restrict__ em, float* __restrict__ vol)
{
  const int b     = blockIdx.z;
  const int mtile = blockIdx.y;
  const int ntile = blockIdx.x;
  const int lane  = threadIdx.x & 31;
  const int half  = lane >> 4;
  const int nl    = lane & 15;

  const bf16_t* e = em + (size_t)b * 16 * 256;

  v16bf a, bb;
  const int m = mtile * 16 + nl;
  #pragma unroll
  for (int v = 0; v < 4; ++v) {                 // kk = half*8 + 2v in [0,15]
    const int kk = (half << 3) + (v << 1);
    a[2 * v]     = e[(kk)     * 256 + m];
    a[2 * v + 1] = e[(kk + 1) * 256 + m];
  }
  #pragma unroll
  for (int v = 4; v < 8; ++v) {                 // kk >= 16: zero pad
    a[2 * v]     = (bf16_t)0.f;
    a[2 * v + 1] = (bf16_t)0.f;
  }
  const int n = ntile * 16 + nl;
  #pragma unroll
  for (int v = 0; v < 8; ++v) {                 // v>=4 -> kk>=16 -> zeros
    const int kk = ((half << 3) + (v << 1)) & 15;
    const bf16_t x0 = e[(kk)     * 256 + n];
    const bf16_t x1 = e[(kk + 1) * 256 + n];
    bb[2 * v]     = (v >= 4) ? (bf16_t)0.f : x0;
    bb[2 * v + 1] = (v >= 4) ? (bf16_t)0.f : x1;
  }

  v8f c = {};
  c = __builtin_amdgcn_wmma_f32_16x16x32_bf16(false, a, false, bb,
                                              (short)0, c, false, false);

  #pragma unroll
  for (int i = 0; i < 8; ++i) {
    const int ij = mtile * 16 + i + (half << 3);
    const int hw = ntile * 16 + nl;
    const float x = c[i] * 0.25f;                  // 1/sqrt(EMBED=16)
    vol[((size_t)b * 256 + ij) * 256 + hw] =
        __builtin_amdgcn_rcpf(1.f + __expf(-x));   // v_exp + v_rcp
  }
}

// =====================================================================
// Host side
// =====================================================================
struct BnP { const float *bias, *mean, *scale, *var; };
struct BlkP { BnP bn1, bn2, bns; const float *w1, *w2, *ws; bool down; };

static void launch_conv(hipStream_t s, const void* in, const float* w,
                        const BnP* bn, const float* bias_only,
                        const bf16_t* resid, bf16_t* out,
                        int B, int Cin, int H, int W, int Cout,
                        int kh, int kw, int stride, int pad, int relu)
{
  const int Ho = (H + 2 * pad - kh) / stride + 1;
  const int Wo = (W + 2 * pad - kw) / stride + 1;
  dim3 grid((Ho * Wo) / 64, (Cout + 31) / 32, B);
  const float* bb = bn ? bn->bias  : nullptr;
  const float* bm = bn ? bn->mean  : nullptr;
  const float* bs = bn ? bn->scale : nullptr;
  const float* bv = bn ? bn->var   : nullptr;
  if (kh == 7) {      // stem reads the f32 image
    conv_bn_wmma<7, 7, 2, 3, float><<<grid, 128, 0, s>>>((const float*)in, w,
        bb, bm, bs, bv, bias_only, resid, out, Cin, H, W, Cout, Ho, Wo, relu);
  } else if (kh == 3 && stride == 1) {
    conv_bn_wmma<3, 3, 1, 1, bf16_t><<<grid, 128, 0, s>>>((const bf16_t*)in, w,
        bb, bm, bs, bv, bias_only, resid, out, Cin, H, W, Cout, Ho, Wo, relu);
  } else if (kh == 3 && stride == 2) {
    conv_bn_wmma<3, 3, 2, 1, bf16_t><<<grid, 128, 0, s>>>((const bf16_t*)in, w,
        bb, bm, bs, bv, bias_only, resid, out, Cin, H, W, Cout, Ho, Wo, relu);
  } else if (kh == 1 && stride == 1) {
    conv_bn_wmma<1, 1, 1, 0, bf16_t><<<grid, 128, 0, s>>>((const bf16_t*)in, w,
        bb, bm, bs, bv, bias_only, resid, out, Cin, H, W, Cout, Ho, Wo, relu);
  } else {
    conv_bn_wmma<1, 1, 2, 0, bf16_t><<<grid, 128, 0, s>>>((const bf16_t*)in, w,
        bb, bm, bs, bv, bias_only, resid, out, Cin, H, W, Cout, Ho, Wo, relu);
  }
}

extern "C" void kernel_launch(void* const* d_in, const int* in_sizes, int n_in,
                              void* d_out, int out_size, void* d_ws, size_t ws_size,
                              hipStream_t stream)
{
  // ---------- input unpack ----------
  // Assumed flattening: top-level dict order {img, w, h, params}; params is a
  // pytree flattened with sorted keys: embed_b, embed_w, head_w, layer1..4
  // (each [block0, block1]; block keys sorted: bn1, bn2, (bns), w1, w2, (ws);
  // bn keys sorted: bias, mean, scale, var), stem_bn, stem_w.
  int ii = 0;
  const float* img = (const float*)d_in[ii++];
  ii++;  // 'w' scalar (spatial dims hardcoded to 256)
  ii++;  // 'h' scalar
  auto nextf  = [&]() { return (const float*)d_in[ii++]; };
  auto readBn = [&]() { BnP p; p.bias = nextf(); p.mean = nextf();
                        p.scale = nextf(); p.var = nextf(); return p; };
  auto readBlk = [&](bool down) {
    BlkP q{}; q.down = down;
    q.bn1 = readBn(); q.bn2 = readBn();
    if (down) q.bns = readBn();
    q.w1 = nextf(); q.w2 = nextf();
    if (down) q.ws = nextf();
    return q;
  };
  const float* embed_b = nextf();
  const float* embed_w = nextf();
  const float* head_w  = nextf();
  BlkP l1[2] = { readBlk(false), readBlk(false) };
  BlkP l2[2] = { readBlk(true),  readBlk(false) };
  BlkP l3[2] = { readBlk(true),  readBlk(false) };
  BlkP l4[2] = { readBlk(true),  readBlk(false) };
  BnP  stem_bn = readBn();
  const float* stem_w = nextf();
  (void)n_in; (void)out_size;

  const int B = in_sizes[0] / (3 * 256 * 256);   // 128

  // ---------- workspace bump allocator (bf16 activations) ----------
  size_t off = 0;
  auto alloc = [&](size_t nelem) {
    bf16_t* p = (bf16_t*)((char*)d_ws + off);
    off += nelem * sizeof(bf16_t);
    off = (off + 255) & ~(size_t)255;
    return p;
  };
  bf16_t* stem_out = alloc((size_t)B * 64 * 128 * 128);
  bf16_t* pool_out = alloc((size_t)B * 64 * 64 * 64);
  bf16_t* tA       = alloc((size_t)B * 64 * 64 * 64);
  bf16_t* tB       = alloc((size_t)B * 64 * 64 * 64);
  bf16_t* uA       = alloc((size_t)B * 128 * 32 * 32);
  bf16_t* uB       = alloc((size_t)B * 128 * 32 * 32);
  bf16_t* uC       = alloc((size_t)B * 128 * 32 * 32);
  bf16_t* vA       = alloc((size_t)B * 256 * 16 * 16);
  bf16_t* vB       = alloc((size_t)B * 256 * 16 * 16);
  bf16_t* vC       = alloc((size_t)B * 256 * 16 * 16);
  bf16_t* wA       = alloc((size_t)B * 512 * 8 * 8);
  bf16_t* wB       = alloc((size_t)B * 512 * 8 * 8);
  bf16_t* wC       = alloc((size_t)B * 512 * 8 * 8);
  float*  pv       = (float*)alloc((size_t)B * 512 * 2);     // f32
  bf16_t* em       = alloc((size_t)B * 16 * 256);
  (void)ws_size;

  float* pred = (float*)d_out;
  float* vol  = pred + (size_t)B * 2;

  // ---------- stem: conv7x7 s2 + BN + ReLU, then maxpool3x3 s2 ----------
  launch_conv(stream, img, stem_w, &stem_bn, nullptr, nullptr, stem_out,
              B, 3, 256, 256, 64, 7, 7, 2, 3, 1);
  {
    const size_t total = (size_t)B * 64 * 64 * 64;
    maxpool3x3s2<<<dim3((unsigned)((total + 255) / 256)), 256, 0, stream>>>(
        stem_out, pool_out, B, 64, 128, 128, 64, 64);
  }

  // ---------- layer1 (64ch, 64x64) ----------
  launch_conv(stream, pool_out, l1[0].w1, &l1[0].bn1, nullptr, nullptr, tA,
              B, 64, 64, 64, 64, 3, 3, 1, 1, 1);
  launch_conv(stream, tA, l1[0].w2, &l1[0].bn2, nullptr, pool_out, tB,
              B, 64, 64, 64, 64, 3, 3, 1, 1, 1);
  launch_conv(stream, tB, l1[1].w1, &l1[1].bn1, nullptr, nullptr, tA,
              B, 64, 64, 64, 64, 3, 3, 1, 1, 1);
  launch_conv(stream, tA, l1[1].w2, &l1[1].bn2, nullptr, tB, pool_out,
              B, 64, 64, 64, 64, 3, 3, 1, 1, 1);

  // ---------- layer2 (64->128, s2, 32x32) ----------
  launch_conv(stream, pool_out, l2[0].w1, &l2[0].bn1, nullptr, nullptr, uA,
              B, 64, 64, 64, 128, 3, 3, 2, 1, 1);
  launch_conv(stream, pool_out, l2[0].ws, &l2[0].bns, nullptr, nullptr, uB,
              B, 64, 64, 64, 128, 1, 1, 2, 0, 0);
  launch_conv(stream, uA, l2[0].w2, &l2[0].bn2, nullptr, uB, uC,
              B, 128, 32, 32, 128, 3, 3, 1, 1, 1);
  launch_conv(stream, uC, l2[1].w1, &l2[1].bn1, nullptr, nullptr, uA,
              B, 128, 32, 32, 128, 3, 3, 1, 1, 1);
  launch_conv(stream, uA, l2[1].w2, &l2[1].bn2, nullptr, uC, uB,
              B, 128, 32, 32, 128, 3, 3, 1, 1, 1);

  // ---------- layer3 (128->256, s2, 16x16) -> source_feature ----------
  launch_conv(stream, uB, l3[0].w1, &l3[0].bn1, nullptr, nullptr, vA,
              B, 128, 32, 32, 256, 3, 3, 2, 1, 1);
  launch_conv(stream, uB, l3[0].ws, &l3[0].bns, nullptr, nullptr, vB,
              B, 128, 32, 32, 256, 1, 1, 2, 0, 0);
  launch_conv(stream, vA, l3[0].w2, &l3[0].bn2, nullptr, vB, vC,
              B, 256, 16, 16, 256, 3, 3, 1, 1, 1);
  launch_conv(stream, vC, l3[1].w1, &l3[1].bn1, nullptr, nullptr, vA,
              B, 256, 16, 16, 256, 3, 3, 1, 1, 1);
  launch_conv(stream, vA, l3[1].w2, &l3[1].bn2, nullptr, vC, vB,
              B, 256, 16, 16, 256, 3, 3, 1, 1, 1);   // vB = source_feature

  // ---------- layer4 (256->512, s2, 8x8) ----------
  launch_conv(stream, vB, l4[0].w1, &l4[0].bn1, nullptr, nullptr, wA,
              B, 256, 16, 16, 512, 3, 3, 2, 1, 1);
  launch_conv(stream, vB, l4[0].ws, &l4[0].bns, nullptr, nullptr, wB,
              B, 256, 16, 16, 512, 1, 1, 2, 0, 0);
  launch_conv(stream, wA, l4[0].w2, &l4[0].bn2, nullptr, wB, wC,
              B, 512, 8, 8, 512, 3, 3, 1, 1, 1);
  launch_conv(stream, wC, l4[1].w1, &l4[1].bn1, nullptr, nullptr, wA,
              B, 512, 8, 8, 512, 3, 3, 1, 1, 1);
  launch_conv(stream, wA, l4[1].w2, &l4[1].bn2, nullptr, wC, wB,
              B, 512, 8, 8, 512, 3, 3, 1, 1, 1);

  // ---------- prediction head ----------
  gap8x8<<<dim3((B * 512 + 255) / 256), 256, 0, stream>>>(wB, pv, B * 512);
  head2<<<dim3((B * 2 + 255) / 256), 256, 0, stream>>>(pv, head_w, pred, B);

  // ---------- embedding (1x1 conv, bias only) + consistency volume ----------
  launch_conv(stream, vB, embed_w, nullptr, embed_b, nullptr, em,
              B, 256, 16, 16, 16, 1, 1, 1, 0, 0);
  corr_sigmoid_wmma<<<dim3(16, 16, B), 32, 0, stream>>>(em, vol);
}
